// Model_25451976196690
// MI455X (gfx1250) — matrix operand order repaired and checked by reading, not compiled, not run on hardware
//
#include <hip/hip_runtime.h>
#include <hip/hip_bf16.h>

// ---------------- types ----------------
typedef _Float16 h16;
typedef __attribute__((ext_vector_type(8)))  h16   v8h;
typedef __attribute__((ext_vector_type(16))) h16   v16h;
typedef __attribute__((ext_vector_type(8)))  float v8f;
typedef int v4i __attribute__((vector_size(16)));

union Frag { v16h v; v8h h[2]; };

__device__ __forceinline__ v8f vzero8() {
  v8f z;
#pragma unroll
  for (int i = 0; i < 8; ++i) z[i] = 0.0f;
  return z;
}

__device__ __forceinline__ v8f wmma_f16(v16h a, v16h b, v8f c) {
  // D = A(16x32) * B(32x16) + C, f32 accumulate
  return __builtin_amdgcn_wmma_f32_16x16x32_f16(false, a, false, b, (short)0, c, false, false);
}

// ---------------- async global->LDS copy (gfx1250), guarded ----------------
#define AS1 __attribute__((address_space(1)))
#define AS3 __attribute__((address_space(3)))

#if defined(__has_builtin)
#if __has_builtin(__builtin_amdgcn_global_load_async_to_lds_b128)
#define HAVE_ASYNC_LDS 1
#else
#define HAVE_ASYNC_LDS 0
#endif
#else
#define HAVE_ASYNC_LDS 0
#endif

// copy 16 bytes global -> LDS (async if hardware path available)
__device__ __forceinline__ void async_copy16(const h16* g, h16* l) {
#if HAVE_ASYNC_LDS
  __builtin_amdgcn_global_load_async_to_lds_b128(
      (AS1 v4i*)(unsigned long long)(const void*)g,
      (AS3 v4i*)(unsigned long long)(unsigned)(unsigned long long)(void*)l,
      0, 0);
#else
  *(v8h*)l = *(const v8h*)g;
#endif
}

__device__ __forceinline__ void wait_async0() {
#if HAVE_ASYNC_LDS
#if defined(__has_builtin)
#if __has_builtin(__builtin_amdgcn_s_wait_asynccnt)
  __builtin_amdgcn_s_wait_asynccnt(0);
#else
  asm volatile("s_wait_asynccnt 0" ::: "memory");
#endif
#else
  asm volatile("s_wait_asynccnt 0" ::: "memory");
#endif
#endif
}

// ---------------- model dims ----------------
#define BB    128
#define NTOK  68
#define BN    (BB * NTOK)     // 8704
#define DIM   512
#define NHEAD 8
#define NRULE 8
#define HD    64
#define QKD   32
#define NPAT  64
#define BP    (BB * NPAT)     // 8192
#define PFP   64              // patch features padded 48 -> 64
#define DFF   1024
#define QKVC  1536            // q|k|v concat
#define LDQKV 1536
#define NPAD  80              // 68 tokens padded to 80 (5 x 16)
#define KPAD  96              // K padding for S@V (3 x 32)
#define KCH   256             // GEMM K-chunk staged in LDS (64 x 256 f16 = 32KB)

// =====================================================================
// Generic WMMA GEMM:  C[M,N] = act(A[M,K] @ Wt^T + bias)
//   A  : f16 row-major, ld = lda
//   Wt : f16, N x K row-major (transposed weight)
//   Weight tile (shared by all 8 waves) is staged in LDS via async copy;
//   A fragments are double-buffered global b128 loads.
//   mode 0: write Cf (f32) and/or Ch (f16)
//   mode 1: Cf += result  (residual accumulate, in-place)
//   mode 2: embed scatter: row -> (row/64)*68 + 4 + row%64, add pe[row%64][col]
// Grid: (M/128, N/64); 256 threads = 8 waves; wave -> 16x64 strip.
// =====================================================================
__global__ __launch_bounds__(256) void gemm_wmma(
    const h16* __restrict__ A, int lda,
    const h16* __restrict__ Wt,
    const float* __restrict__ bias,
    float* __restrict__ Cf, h16* __restrict__ Ch, int ldc,
    int K, int relu, int mode, const float* __restrict__ pe)
{
  __shared__ h16 bs[64 * KCH];            // staged weight tile

  const int lane = threadIdx.x & 31;
  const int wv   = threadIdx.x >> 5;
  const int hs   = lane >> 4;             // half-select
  const int mr   = lane & 15;
  const int m0   = (blockIdx.x * 8 + wv) * 16;
  const int n0   = blockIdx.y * 64;

  v8f acc[4];
#pragma unroll
  for (int t = 0; t < 4; ++t) acc[t] = vzero8();

  const h16* __restrict__ arow  = A + (size_t)(m0 + mr) * lda;
  const h16* __restrict__ wrow0 = Wt + (size_t)n0 * K;

  for (int kc = 0; kc < K; kc += KCH) {
    const int klen = (K - kc) < KCH ? (K - kc) : KCH;   // 64 or 256, mult of 32
    // ---- stage 64 x klen weight tile into LDS (async) ----
    const int cpr    = klen >> 3;          // 16B chunks per row
    const int chunks = 64 * cpr;
    for (int c = threadIdx.x; c < chunks; c += 256) {
      const int row = c / cpr;
      const int off = (c - row * cpr) * 8;
      async_copy16(wrow0 + (size_t)row * K + kc + off, &bs[row * klen + off]);
    }
    wait_async0();
    __syncthreads();

    // ---- compute, A double-buffered ----
    Frag a;
    a.h[0] = *(const v8h*)(arow + kc + hs * 16);
    a.h[1] = *(const v8h*)(arow + kc + hs * 16 + 8);
    for (int kk = 0; kk < klen; kk += 32) {
      Frag an;
      const bool more = (kk + 32) < klen;
      if (more) {
        an.h[0] = *(const v8h*)(arow + kc + kk + 32 + hs * 16);
        an.h[1] = *(const v8h*)(arow + kc + kk + 32 + hs * 16 + 8);
      }
#pragma unroll
      for (int t = 0; t < 4; ++t) {
        Frag b;
        const h16* bp = &bs[(t * 16 + mr) * klen + kk + hs * 16];
        b.h[0] = *(const v8h*)(bp);
        b.h[1] = *(const v8h*)(bp + 8);
        acc[t] = wmma_f16(a.v, b.v, acc[t]);
      }
      if (more) a = an;
    }
    __syncthreads();   // protect bs before next stage
  }

#pragma unroll
  for (int t = 0; t < 4; ++t) {
    const int col = n0 + t * 16 + mr;
    const float bi = bias ? bias[col] : 0.0f;
#pragma unroll
    for (int j = 0; j < 8; ++j) {
      const int row = m0 + hs * 8 + j;
      float v = acc[t][j] + bi;
      if (relu) v = v > 0.0f ? v : 0.0f;
      if (mode == 0) {
        if (Cf) Cf[(size_t)row * ldc + col] = v;
        if (Ch) Ch[(size_t)row * ldc + col] = (h16)v;
      } else if (mode == 1) {
        Cf[(size_t)row * ldc + col] += v;
      } else { // embed scatter + positional encoding
        const int p = row & 63;
        const int orow = (row >> 6) * NTOK + 4 + p;
        Cf[(size_t)orow * ldc + col] = v + pe[(size_t)p * ldc + col];
      }
    }
  }
}

// =====================================================================
// LayerNorm over 512 cols -> f16 output. One block (256 thr) per row.
// =====================================================================
__global__ __launch_bounds__(256) void ln_rows(
    const float* __restrict__ X, const float* __restrict__ g,
    const float* __restrict__ b, h16* __restrict__ out)
{
  __shared__ float red[256];
  const int row = blockIdx.x;
  const float* x = X + (size_t)row * DIM;
  const int c0 = threadIdx.x, c1 = threadIdx.x + 256;
  const float a0 = x[c0], a1 = x[c1];

  red[threadIdx.x] = a0 + a1;
  __syncthreads();
  for (int off = 128; off > 0; off >>= 1) {
    if (threadIdx.x < off) red[threadIdx.x] += red[threadIdx.x + off];
    __syncthreads();
  }
  const float mean = red[0] * (1.0f / DIM);
  __syncthreads();

  const float d0 = a0 - mean, d1 = a1 - mean;
  red[threadIdx.x] = d0 * d0 + d1 * d1;
  __syncthreads();
  for (int off = 128; off > 0; off >>= 1) {
    if (threadIdx.x < off) red[threadIdx.x] += red[threadIdx.x + off];
    __syncthreads();
  }
  const float inv = rsqrtf(red[0] * (1.0f / DIM) + 1e-5f);
  out[(size_t)row * DIM + c0] = (h16)(d0 * inv * g[c0] + b[c0]);
  out[(size_t)row * DIM + c1] = (h16)(d1 * inv * g[c1] + b[c1]);
}

// =====================================================================
// Attention stage 1: per (b,h) block -> softmax(Q K^T) as padded f16 tile.
// Q/K head tiles staged in LDS (async); 25 WMMA tiles over 8 waves.
// Sbuf tile: 80 x 96 f16, zero padded.
// =====================================================================
__global__ __launch_bounds__(256) void attn_scores(
    const h16* __restrict__ QKV, h16* __restrict__ Sbuf)
{
  __shared__ float Sf[NPAD][NPAD];
  __shared__ h16 Qs[NTOK][HD];
  __shared__ h16 Ks[NTOK][HD];

  const int bh = blockIdx.x;            // b*8 + h
  const int b  = bh >> 3;
  const int h  = bh & 7;
  const int lane = threadIdx.x & 31;
  const int wv   = threadIdx.x >> 5;
  const int hs = lane >> 4, mr = lane & 15;

  const h16* Qb = QKV + (size_t)(b * NTOK) * LDQKV + h * HD;
  const h16* Kb = Qb + DIM;

  // stage Q and K head tiles (68 x 64 f16 each) into LDS
  const int QCH = (NTOK * HD) / 8;      // 16B chunks per matrix (544)
  for (int c = threadIdx.x; c < 2 * QCH; c += 256) {
    const int m  = (c >= QCH);
    const int cc = c - m * QCH;
    const int row = cc >> 3;
    const int off = (cc & 7) * 8;
    const h16* src = (m ? Kb : Qb) + (size_t)row * LDQKV + off;
    h16* dst = (m ? &Ks[0][0] : &Qs[0][0]) + row * HD + off;
    async_copy16(src, dst);
  }
  wait_async0();
  __syncthreads();

  for (int tix = wv; tix < 25; tix += 8) {
    const int m0 = (tix / 5) * 16, n0 = (tix % 5) * 16;
    int qr = m0 + mr; if (qr > NTOK - 1) qr = NTOK - 1;
    int kr = n0 + mr; if (kr > NTOK - 1) kr = NTOK - 1;
    v8f acc = vzero8();
#pragma unroll
    for (int kk = 0; kk < HD; kk += 32) {
      Frag a, bq;
      a.h[0]  = *(const v8h*)(&Qs[qr][kk + hs * 16]);
      a.h[1]  = *(const v8h*)(&Qs[qr][kk + hs * 16 + 8]);
      bq.h[0] = *(const v8h*)(&Ks[kr][kk + hs * 16]);
      bq.h[1] = *(const v8h*)(&Ks[kr][kk + hs * 16 + 8]);
      acc = wmma_f16(a.v, bq.v, acc);
    }
#pragma unroll
    for (int j = 0; j < 8; ++j) Sf[m0 + hs * 8 + j][n0 + mr] = acc[j];
  }
  __syncthreads();

  const int t = threadIdx.x;
  if (t < NPAD) {
    h16* o = Sbuf + (size_t)bh * NPAD * KPAD + (size_t)t * KPAD;
    if (t < NTOK) {
      float mx = -1e30f;
      for (int c = 0; c < NTOK; ++c) mx = fmaxf(mx, Sf[t][c]);
      float s = 0.0f;
      for (int c = 0; c < NTOK; ++c) { float e = __expf(Sf[t][c] - mx); Sf[t][c] = e; s += e; }
      const float inv = 1.0f / s;
      for (int c = 0; c < NTOK; ++c) o[c] = (h16)(Sf[t][c] * inv);
      for (int c = NTOK; c < KPAD; ++c) o[c] = (h16)0.0f;
    } else {
      for (int c = 0; c < KPAD; ++c) o[c] = (h16)0.0f;
    }
  }
}

// =====================================================================
// Repack V into [b][col][tok] f16 with tok zero-padded to KPAD, so S@V
// B-fragments are two contiguous b128 loads.
// =====================================================================
__global__ void repack_v(const h16* __restrict__ QKV, h16* __restrict__ Vt)
{
  const int i = blockIdx.x * 256 + threadIdx.x;
  if (i >= BB * DIM * KPAD) return;
  const int tok = i % KPAD;
  const int col = (i / KPAD) % DIM;
  const int b   = i / (KPAD * DIM);
  h16 v = (h16)0.0f;
  if (tok < NTOK) v = QKV[(size_t)(b * NTOK + tok) * LDQKV + 2 * DIM + col];
  Vt[i] = v;
}

// =====================================================================
// Attention stage 2: per (b,h,token-tile) block.
//   S tile (16x96) staged in LDS (async, shared by all 8 waves);
//   out = S @ V via WMMA into LDS; rule-softmax 'comp' and contraction
//   -> O[:, h*64 : h*64+64] in f16.
// =====================================================================
__global__ __launch_bounds__(256) void attn_out(
    const h16* __restrict__ Sbuf, const h16* __restrict__ Vt,
    const float* __restrict__ qvd, const float* __restrict__ score_w,
    h16* __restrict__ O)
{
  __shared__ float outS[16][DIM];
  __shared__ float Lg[16][NRULE];
  __shared__ h16 Ss[16][KPAD];

  const int blk = blockIdx.x;           // b*40 + h*5 + mt
  const int mt  = blk % 5;
  const int bh  = blk / 5;
  const int h   = bh & 7;
  const int b   = bh >> 3;
  const int m0  = mt * 16;
  const int lane = threadIdx.x & 31;
  const int wv   = threadIdx.x >> 5;
  const int hs = lane >> 4, mr = lane & 15;

  const h16* Sb = Sbuf + (size_t)bh * NPAD * KPAD;

  // stage the 16 x 96 score tile (rows m0..m0+15, zero rows included)
  if (threadIdx.x < 16 * 12) {
    const int row = threadIdx.x / 12;
    const int off = (threadIdx.x - row * 12) * 8;
    async_copy16(Sb + (size_t)(m0 + row) * KPAD + off, &Ss[row][off]);
  }
  wait_async0();
  __syncthreads();

  v8f acc[4];
#pragma unroll
  for (int t = 0; t < 4; ++t) acc[t] = vzero8();

#pragma unroll
  for (int kk = 0; kk < KPAD; kk += 32) {
    Frag a;
    a.h[0] = *(const v8h*)(&Ss[mr][kk + hs * 16]);
    a.h[1] = *(const v8h*)(&Ss[mr][kk + hs * 16 + 8]);
#pragma unroll
    for (int t = 0; t < 4; ++t) {
      const int col = wv * 64 + t * 16 + mr;
      Frag bf;
      const h16* bp = Vt + ((size_t)b * DIM + col) * KPAD + kk + hs * 16;
      bf.h[0] = *(const v8h*)(bp);
      bf.h[1] = *(const v8h*)(bp + 8);
      acc[t] = wmma_f16(a.v, bf.v, acc[t]);
    }
  }
#pragma unroll
  for (int t = 0; t < 4; ++t)
#pragma unroll
    for (int j = 0; j < 8; ++j)
      outS[hs * 8 + j][wv * 64 + t * 16 + mr] = acc[t][j];
  __syncthreads();

  // comp logits: qv.score_w (precomputed) + out.score_w[32:]
  if (threadIdx.x < 128) {
    const int lr = threadIdx.x >> 3, r = threadIdx.x & 7;
    const int tok = m0 + lr;
    float s = 0.0f;
    if (tok < NTOK) {
      s = qvd[((size_t)(b * NTOK + tok)) * NHEAD + h];
      for (int d = 0; d < HD; ++d) s += outS[lr][r * HD + d] * score_w[QKD + d];
    }
    Lg[lr][r] = s;
  }
  __syncthreads();

  if (threadIdx.x < 16) {
    const int lr = threadIdx.x;
    float mx = Lg[lr][0];
    for (int r = 1; r < NRULE; ++r) mx = fmaxf(mx, Lg[lr][r]);
    float s = 0.0f;
    for (int r = 0; r < NRULE; ++r) { float e = __expf(Lg[lr][r] - mx); Lg[lr][r] = e; s += e; }
    const float inv = 1.0f / s;
    for (int r = 0; r < NRULE; ++r) Lg[lr][r] *= inv;
  }
  __syncthreads();

  for (int idx = threadIdx.x; idx < 16 * HD; idx += 256) {
    const int lr = idx >> 6, d = idx & 63;
    const int tok = m0 + lr;
    if (tok < NTOK) {
      float s = 0.0f;
#pragma unroll
      for (int r = 0; r < NRULE; ++r) s += Lg[lr][r] * outS[lr][r * HD + d];
      O[((size_t)(b * NTOK + tok)) * DIM + h * HD + d] = (h16)s;
    }
  }
}

// =====================================================================
// Small helper kernels
// =====================================================================
__global__ void convert_wt(const float* __restrict__ W, h16* __restrict__ Wt,
                           int K, int Ncols, float scale)
{
  const int i = blockIdx.x * 256 + threadIdx.x;
  if (i >= K * Ncols) return;
  const int n = i / K, k = i % K;
  Wt[i] = (h16)(W[(size_t)k * Ncols + n] * scale);
}

__global__ void convert_wt_pad(const float* __restrict__ W, h16* __restrict__ Wt,
                               int Kr, int Kp, int Ncols)
{
  const int i = blockIdx.x * 256 + threadIdx.x;
  if (i >= Kp * Ncols) return;
  const int n = i / Kp, k = i % Kp;
  Wt[i] = (k < Kr) ? (h16)W[(size_t)k * Ncols + n] : (h16)0.0f;
}

__global__ void scale_copy(const float* __restrict__ in, float* __restrict__ out,
                           int n, float scale)
{
  const int i = blockIdx.x * 256 + threadIdx.x;
  if (i < n) out[i] = in[i] * scale;
}

__global__ void build_patches(const float* __restrict__ img, h16* __restrict__ Ph)
{
  const int i = blockIdx.x * 256 + threadIdx.x;
  if (i >= BP * PFP) return;
  const int f = i & 63, row = i >> 6;
  const int b = row >> 6, patch = row & 63;
  float v = 0.0f;
  if (f < 48) {
    const int py = patch >> 3, px = patch & 7;
    const int iy = f / 12, rem = f % 12, ix = rem / 3, c = rem % 3;
    const int y = py * 4 + iy, x = px * 4 + ix;
    v = img[(((size_t)b * 3 + c) * 32 + y) * 32 + x];
  }
  Ph[i] = (h16)v;
}

__global__ __launch_bounds__(512) void cls_embed(
    const float* __restrict__ cls, const float* __restrict__ w1,
    const float* __restrict__ b1, const float* __restrict__ w2,
    const float* __restrict__ b2, float* __restrict__ out)
{
  __shared__ float hid[DIM];
  const int t = blockIdx.x, j = threadIdx.x;
  float s = b1[j];
  for (int f = 0; f < 48; ++f) s += cls[t * 48 + f] * w1[(size_t)f * DIM + j];
  hid[j] = s > 0.0f ? s : 0.0f;
  __syncthreads();
  float o = b2[j];
  for (int i = 0; i < DIM; ++i) o += hid[i] * w2[(size_t)i * DIM + j];
  out[t * DIM + j] = o;
}

__global__ void scatter_cls(const float* __restrict__ clsE, float* __restrict__ X)
{
  const int i = blockIdx.x * 256 + threadIdx.x;
  if (i >= BB * 4 * DIM) return;
  const int c = i & 511, r = i >> 9;      // r = b*4 + t
  const int b = r >> 2, t = r & 3;
  X[((size_t)(b * NTOK + t)) * DIM + c] = clsE[(t << 9) + c];
}

__global__ void gather_cls(const float* __restrict__ X, h16* __restrict__ Xc)
{
  const int i = blockIdx.x * 256 + threadIdx.x;
  if (i >= BB * 4 * DIM) return;
  const int c = i & 511, r = i >> 9;      // r = b*4 + t
  const int b = r >> 2, t = r & 3;
  Xc[i] = (h16)X[((size_t)(b * NTOK + t)) * DIM + c];
}

__global__ void qv_dot(const float* __restrict__ QVf, const float* __restrict__ score_w,
                       const float* __restrict__ score_b, float* __restrict__ qvd)
{
  const int i = blockIdx.x * 256 + threadIdx.x;
  if (i >= BN * NHEAD) return;
  const int row = i >> 3, h = i & 7;
  const float* q = QVf + (size_t)row * (QKD * NHEAD) + h * QKD;
  float s = score_b[0];
#pragma unroll
  for (int j = 0; j < QKD; ++j) s += q[j] * score_w[j];
  qvd[i] = s;
}

__global__ void heads(const float* __restrict__ Y,
                      const float* __restrict__ cw, const float* __restrict__ cb,
                      const float* __restrict__ fw, const float* __restrict__ fb,
                      const float* __restrict__ sw, const float* __restrict__ sb,
                      const float* __restrict__ ew, const float* __restrict__ eb,
                      float* __restrict__ out)
{
  const int i = blockIdx.x * 256 + threadIdx.x;
  if (i >= BB * 31) return;
  const int b = i / 31, o = i % 31;
  int task, oo, stride; const float* W; float bias; float* dst;
  if (o < 10)      { task = 0; oo = o;      W = cw; bias = cb[oo]; dst = out + b * 10 + oo;        stride = 10; }
  else if (o < 20) { task = 1; oo = o - 10; W = fw; bias = fb[oo]; dst = out + 1280 + b * 10 + oo; stride = 10; }
  else if (o < 30) { task = 2; oo = o - 20; W = sw; bias = sb[oo]; dst = out + 2560 + b * 10 + oo; stride = 10; }
  else             { task = 3; oo = 0;      W = ew; bias = eb[0];  dst = out + 3840 + b;           stride = 1;  }
  const float* y = Y + (size_t)(b * 4 + task) * 128;
  float s = bias;
  for (int j = 0; j < 128; ++j) s += y[j] * W[j * stride + oo];
  *dst = s;
}

// =====================================================================
// Host launcher
// =====================================================================
extern "C" void kernel_launch(void* const* d_in, const int* in_sizes, int n_in,
                              void* d_out, int out_size, void* d_ws, size_t ws_size,
                              hipStream_t stream)
{
  (void)in_sizes; (void)n_in; (void)out_size; (void)ws_size;

  const float* img      = (const float*)d_in[0];
  const float* cls_tok  = (const float*)d_in[1];
  const float* emb_w1   = (const float*)d_in[2];
  const float* emb_b1   = (const float*)d_in[3];
  const float* emb_w2   = (const float*)d_in[4];
  const float* emb_b2   = (const float*)d_in[5];
  const float* pe       = (const float*)d_in[6];
  const float* ln1_g    = (const float*)d_in[7];
  const float* ln1_b    = (const float*)d_in[8];
  const float* ln2_g    = (const float*)d_in[9];
  const float* ln2_b    = (const float*)d_in[10];
  const float* q_w      = (const float*)d_in[11];
  const float* q_b      = (const float*)d_in[12];
  const float* k_w      = (const float*)d_in[13];
  const float* k_b      = (const float*)d_in[14];
  const float* v_w      = (const float*)d_in[15];
  const float* v_b      = (const float*)d_in[16];
  const float* qv_w     = (const float*)d_in[17];
  const float* qv_b     = (const float*)d_in[18];
  const float* score_w  = (const float*)d_in[19];
  const float* score_b  = (const float*)d_in[20];
  const float* final_w  = (const float*)d_in[21];
  const float* final_b  = (const float*)d_in[22];
  const float* res_w1   = (const float*)d_in[23];
  const float* res_b1   = (const float*)d_in[24];
  const float* res_w2   = (const float*)d_in[25];
  const float* res_b2   = (const float*)d_in[26];
  const float* out_w    = (const float*)d_in[27];
  const float* out_b    = (const float*)d_in[28];
  const float* cifar_w  = (const float*)d_in[29];
  const float* cifar_b  = (const float*)d_in[30];
  const float* fmnist_w = (const float*)d_in[31];
  const float* fmnist_b = (const float*)d_in[32];
  const float* svhn_w   = (const float*)d_in[33];
  const float* svhn_b   = (const float*)d_in[34];
  const float* eq_w     = (const float*)d_in[35];
  const float* eq_b     = (const float*)d_in[36];
  float* outp = (float*)d_out;

  // ---- workspace carve (256B aligned bump allocator) ----
  char* wp = (char*)d_ws;
  auto alloc = [&](size_t bytes) -> void* {
    void* r = (void*)wp;
    wp += (bytes + 255) & ~(size_t)255;
    return r;
  };
  h16*  wt_emb1  = (h16*)alloc((size_t)DIM * PFP * 2);
  h16*  wt_emb2  = (h16*)alloc((size_t)DIM * DIM * 2);
  h16*  wt_qkv   = (h16*)alloc((size_t)QKVC * DIM * 2);
  float* b_qkv   = (float*)alloc((size_t)QKVC * 4);
  h16*  wt_qv    = (h16*)alloc((size_t)(QKD * NHEAD) * DIM * 2);
  h16*  wt_fin   = (h16*)alloc((size_t)DIM * DIM * 2);
  h16*  wt_r1    = (h16*)alloc((size_t)DFF * DIM * 2);
  h16*  wt_r2    = (h16*)alloc((size_t)DIM * DFF * 2);
  h16*  wt_out   = (h16*)alloc((size_t)128 * DIM * 2);
  h16*  Ph       = (h16*)alloc((size_t)BP * PFP * 2);
  h16*  T1h      = (h16*)alloc((size_t)BP * DIM * 2);
  float* X       = (float*)alloc((size_t)BN * DIM * 4);
  h16*  XNh      = (h16*)alloc((size_t)BN * DIM * 2);
  h16*  QKVh     = (h16*)alloc((size_t)BN * QKVC * 2);
  float* QVf     = (float*)alloc((size_t)BN * (QKD * NHEAD) * 4);
  float* qvd     = (float*)alloc((size_t)BN * NHEAD * 4);
  h16*  Sbuf     = (h16*)alloc((size_t)BB * NHEAD * NPAD * KPAD * 2);
  h16*  Vt       = (h16*)alloc((size_t)BB * DIM * KPAD * 2);
  h16*  Obuf     = (h16*)alloc((size_t)BN * DIM * 2);
  h16*  R1h      = (h16*)alloc((size_t)BN * DFF * 2);
  float* clsE    = (float*)alloc((size_t)4 * DIM * 4);
  h16*  Xclsh    = (h16*)alloc((size_t)BB * 4 * DIM * 2);
  float* Y       = (float*)alloc((size_t)BB * 4 * 128 * 4);

  auto cdiv = [](int a, int b) { return (a + b - 1) / b; };
  const float qscale = 0.125f; // 1/sqrt(HD)

  // ---- weight prep (f16, transposed to N x K) ----
  convert_wt_pad<<<cdiv(DIM * PFP, 256), 256, 0, stream>>>(emb_w1, wt_emb1, 48, PFP, DIM);
  convert_wt<<<cdiv(DIM * DIM, 256), 256, 0, stream>>>(emb_w2, wt_emb2, DIM, DIM, 1.0f);
  convert_wt<<<cdiv(DIM * DIM, 256), 256, 0, stream>>>(q_w, wt_qkv,                         DIM, DIM, qscale);
  convert_wt<<<cdiv(DIM * DIM, 256), 256, 0, stream>>>(k_w, wt_qkv + (size_t)DIM * DIM,     DIM, DIM, 1.0f);
  convert_wt<<<cdiv(DIM * DIM, 256), 256, 0, stream>>>(v_w, wt_qkv + (size_t)2 * DIM * DIM, DIM, DIM, 1.0f);
  scale_copy<<<cdiv(DIM, 256), 256, 0, stream>>>(q_b, b_qkv,           DIM, qscale);
  scale_copy<<<cdiv(DIM, 256), 256, 0, stream>>>(k_b, b_qkv + DIM,     DIM, 1.0f);
  scale_copy<<<cdiv(DIM, 256), 256, 0, stream>>>(v_b, b_qkv + 2 * DIM, DIM, 1.0f);
  convert_wt<<<cdiv(256 * DIM, 256), 256, 0, stream>>>(qv_w, wt_qv, DIM, 256, 1.0f);
  convert_wt<<<cdiv(DIM * DIM, 256), 256, 0, stream>>>(final_w, wt_fin, DIM, DIM, 1.0f);
  convert_wt<<<cdiv(DFF * DIM, 256), 256, 0, stream>>>(res_w1, wt_r1, DIM, DFF, 1.0f);
  convert_wt<<<cdiv(DIM * DFF, 256), 256, 0, stream>>>(res_w2, wt_r2, DFF, DIM, 1.0f);
  convert_wt<<<cdiv(128 * DIM, 256), 256, 0, stream>>>(out_w, wt_out, DIM, 128, 1.0f);

  // ---- embedding ----
  build_patches<<<cdiv(BP * PFP, 256), 256, 0, stream>>>(img, Ph);
  cls_embed<<<4, 512, 0, stream>>>(cls_tok, emb_w1, emb_b1, emb_w2, emb_b2, clsE);
  scatter_cls<<<cdiv(BB * 4 * DIM, 256), 256, 0, stream>>>(clsE, X);
  // T1 = relu(P @ emb_w1 + b1)  -> f16
  gemm_wmma<<<dim3(BP / 128, DIM / 64), 256, 0, stream>>>(
      Ph, PFP, wt_emb1, emb_b1, nullptr, T1h, DIM, PFP, 1, 0, nullptr);
  // X[patch rows] = T1 @ emb_w2 + b2 + pe   (scatter mode)
  gemm_wmma<<<dim3(BP / 128, DIM / 64), 256, 0, stream>>>(
      T1h, DIM, wt_emb2, emb_b2, X, nullptr, DIM, DIM, 0, 2, pe);

  // ---- 4 recurrent transformer iterations ----
  for (int it = 0; it < 4; ++it) {
    ln_rows<<<BN, 256, 0, stream>>>(X, ln1_g, ln1_b, XNh);
    // fused q|k|v projection (f16 out), q pre-scaled
    gemm_wmma<<<dim3(BN / 128, QKVC / 64), 256, 0, stream>>>(
        XNh, DIM, wt_qkv, b_qkv, nullptr, QKVh, QKVC, DIM, 0, 0, nullptr);
    // qv projection (f32 out)
    gemm_wmma<<<dim3(BN / 128, (QKD * NHEAD) / 64), 256, 0, stream>>>(
        XNh, DIM, wt_qv, qv_b, QVf, nullptr, QKD * NHEAD, DIM, 0, 0, nullptr);
    qv_dot<<<cdiv(BN * NHEAD, 256), 256, 0, stream>>>(QVf, score_w, score_b, qvd);
    repack_v<<<cdiv(BB * DIM * KPAD, 256), 256, 0, stream>>>(QKVh, Vt);
    attn_scores<<<BB * NHEAD, 256, 0, stream>>>(QKVh, Sbuf);
    attn_out<<<BB * NHEAD * 5, 256, 0, stream>>>(Sbuf, Vt, qvd, score_w, Obuf);
    // X += O @ final_w + final_b   (residual)
    gemm_wmma<<<dim3(BN / 128, DIM / 64), 256, 0, stream>>>(
        Obuf, DIM, wt_fin, final_b, X, nullptr, DIM, DIM, 0, 1, nullptr);
    ln_rows<<<BN, 256, 0, stream>>>(X, ln2_g, ln2_b, XNh);
    // R1 = relu(XN @ res_w1 + b1)
    gemm_wmma<<<dim3(BN / 128, DFF / 64), 256, 0, stream>>>(
        XNh, DIM, wt_r1, res_b1, nullptr, R1h, DFF, DIM, 1, 0, nullptr);
    // X += R1 @ res_w2 + b2
    gemm_wmma<<<dim3(BN / 128, DIM / 64), 256, 0, stream>>>(
        R1h, DFF, wt_r2, res_b2, X, nullptr, DIM, DFF, 0, 1, nullptr);
  }

  // ---- output heads ----
  gather_cls<<<cdiv(BB * 4 * DIM, 256), 256, 0, stream>>>(X, Xclsh);
  gemm_wmma<<<dim3((BB * 4) / 128, 128 / 64), 256, 0, stream>>>(
      Xclsh, DIM, wt_out, out_b, Y, nullptr, 128, DIM, 1, 0, nullptr);
  heads<<<cdiv(BB * 31, 256), 256, 0, stream>>>(
      Y, cifar_w, cifar_b, fmnist_w, fmnist_b, svhn_w, svhn_b, eq_w, eq_b, outp);
}
